// GAIN_GloVe_13073880449184
// MI455X (gfx1250) — matrix operand order repaired and checked by reading, not compiled, not run on hardware
//
#include <hip/hip_runtime.h>
#include <math.h>

// ---------------------------------------------------------------------------
// Problem constants (from reference)
// ---------------------------------------------------------------------------
#define GCN      512
#define BANKD    1536
#define NNODE    1024      // BSZ*ENT
#define NEDGE    32768
#define NPAIRTOT 1024      // BSZ*NPAIR
#define KPATH    8
#define PATHROWS 8192      // NPAIRTOT*KPATH
#define FOURGCN  2048
#define QDIM     3072      // 2*BANK
#define FEATDIM  8704
#define HIDDIM   3072
#define OUTDIM   97

typedef __attribute__((ext_vector_type(16))) __bf16 v16bf;
typedef __attribute__((ext_vector_type(8)))  __bf16 v8bf;
typedef __attribute__((ext_vector_type(8)))  float  v8f;

__device__ inline __bf16 f2bf(float f) {
  union { float f; unsigned u; } v; v.f = f;
  unsigned r = (v.u + 0x7FFFu + ((v.u >> 16) & 1u)) >> 16;  // round-nearest-even
  union { unsigned short s; __bf16 b; } o; o.s = (unsigned short)r;
  return o.b;
}

__device__ inline v16bf frag16(const __bf16* p0, const __bf16* p1) {
  v8bf lo = *(const v8bf*)p0;   // ds_load_b128
  v8bf hi = *(const v8bf*)p1;   // ds_load_b128
  return __builtin_shufflevector(lo, hi, 0, 1, 2, 3, 4, 5, 6, 7,
                                 8, 9, 10, 11, 12, 13, 14, 15);
}

// ---------------------------------------------------------------------------
// Fused WMMA GEMM:  C = epilogue( A[M,K](bf16) @ Bt[N,K](bf16)^T )
// Tiles: BM=64, BN=128, BK=32; 256 thr = 8 waves; wave -> m-strip (w&3),
// n-half (w>>2)*64, 4 accumulators => 4 WMMA per K-step.
// Double-buffered LDS staging: global loads for tile k+1 are issued before
// the WMMAs of tile k; the loadcnt wait lands after compute.
// Requires M%64==0, K%32==0. RAGGED=true adds the N-boundary guard.
// ---------------------------------------------------------------------------
#define BM 64
#define BN 128
#define BK 32

template <bool RAGGED>
__global__ __launch_bounds__(256) void gemm_wmma(
    const __bf16* __restrict__ A, const __bf16* __restrict__ Bt,
    const float* __restrict__ bias, const float* __restrict__ rowscale,
    void* __restrict__ Cv, int M, int N, int K, int fuse_tanh, int out_bf16)
{
  __shared__ __align__(16) __bf16 As[2][BM][BK];   // row-major (K contiguous)
  __shared__ __align__(16) __bf16 Bs[2][BN][BK];   // col-major (K contiguous)

  const int tid  = threadIdx.x;
  const int wave = tid >> 5;
  const int lane = tid & 31;
  const int half = lane >> 4;      // 0: lanes 0-15, 1: lanes 16-31
  const int l16  = lane & 15;

  const int m0 = blockIdx.y * BM;
  const int n0 = blockIdx.x * BN;
  const int wm = wave & 3;         // 16-row strip
  const int wn = (wave >> 2) * 64; // 0 or 64 within 128-col tile

  // staging assignments: A: 256 groups of 8; B: 512 groups -> 2 per thread
  const int ar  = tid >> 2, ac = (tid & 3) * 8;
  const int bc  = tid & 127;
  const int brg = (tid >> 7) * 8;  // 0 or 8; second group at +16

  v8f acc[4] = {{}, {}, {}, {}};
  v8bf ldA, ldB0, ldB1;

  auto load_tile = [&](int k0) {
    ldA = *(const v8bf*)&A[(size_t)(m0 + ar) * K + k0 + ac];
    const int gc = n0 + bc;
    if (!RAGGED || gc < N) {
      const __bf16* bp = &Bt[(size_t)gc * K + k0 + brg];
      ldB0 = *(const v8bf*)bp;
      ldB1 = *(const v8bf*)(bp + 16);
    } else {
      ldB0 = (v8bf){};
      ldB1 = (v8bf){};
    }
    if (k0 + BK < K) {  // speculative prefetch of the tile after next
      __builtin_prefetch(&A[(size_t)(m0 + ar) * K + k0 + BK + ac], 0, 0);
      if (!RAGGED || n0 + bc < N)
        __builtin_prefetch(&Bt[(size_t)(n0 + bc) * K + k0 + BK], 0, 0);
    }
  };
  auto store_tile = [&](int buf) {
    *(v8bf*)&As[buf][ar][ac] = ldA;
    *(v8bf*)&Bs[buf][bc][brg] = ldB0;
    *(v8bf*)&Bs[buf][bc][brg + 16] = ldB1;
  };

  const int KT = K / BK;
  load_tile(0);
  store_tile(0);
  __syncthreads();

  for (int kt = 0; kt < KT; ++kt) {
    const int cur = kt & 1;
    if (kt + 1 < KT) load_tile((kt + 1) * BK);   // overlap with compute below

    const int arow = wm * 16 + l16;
    v16bf a = frag16(&As[cur][arow][half * 8], &As[cur][arow][16 + half * 8]);
    v16bf bf[4];
#pragma unroll
    for (int t = 0; t < 4; ++t) {
      const int cb = wn + t * 16 + l16;
      bf[t] = frag16(&Bs[cur][cb][half * 8], &Bs[cur][cb][16 + half * 8]);
    }
#pragma unroll
    for (int t = 0; t < 4; ++t)
      acc[t] = __builtin_amdgcn_wmma_f32_16x16x32_bf16(false, a, false, bf[t],
                                                       (short)0, acc[t],
                                                       false, false);

    if (kt + 1 < KT) store_tile(1 - cur);        // loadcnt wait lands here
    __syncthreads();
  }

  // --- epilogue: C layout VGPR i -> row = i + 8*half, col = l16 ---
  const int rbase = m0 + wm * 16 + half * 8;
#pragma unroll
  for (int t = 0; t < 4; ++t) {
    const int c = n0 + wn + t * 16 + l16;
    if (RAGGED && c >= N) continue;
    const float bv = bias ? bias[c] : 0.0f;
#pragma unroll
    for (int i = 0; i < 8; ++i) {
      int r = rbase + i;
      float s = rowscale ? rowscale[r] : 1.0f;
      float v = acc[t][i] * s + bv;
      if (fuse_tanh) v = tanhf(v);
      if (out_bf16) ((__bf16*)Cv)[(size_t)r * N + c] = f2bf(v);
      else          ((float*)Cv)[(size_t)r * N + c] = v;
    }
  }
}

// ---------------------------------------------------------------------------
// Weight conversion: fp32 W[K,N] -> bf16 Wt[N,K] (transposed, once per call)
// ---------------------------------------------------------------------------
__global__ void convert_transpose(const float* __restrict__ W,
                                  __bf16* __restrict__ Wt, int K, int N)
{
  int i = blockIdx.x * blockDim.x + threadIdx.x;   // output-contiguous
  if (i >= K * N) return;
  int n = i / K, k = i % K;
  Wt[i] = f2bf(W[(size_t)k * N + n]);
}

__global__ void convert_bf(const float* __restrict__ x, __bf16* __restrict__ y, int n)
{
  int i = blockIdx.x * blockDim.x + threadIdx.x;
  if (i < n) y[i] = f2bf(x[i]);
}

// ---------------------------------------------------------------------------
// Sparse aggregation:  agg[dst[e], :] += x[src[e], :]
// ---------------------------------------------------------------------------
__global__ __launch_bounds__(256) void spmm_atomic(
    const float* __restrict__ x, const int* __restrict__ src,
    const int* __restrict__ dst, float* __restrict__ agg)
{
  int e = blockIdx.x >> 1;
  int c = ((blockIdx.x & 1) << 8) + threadIdx.x;
  int s = src[e], d = dst[e];
  atomicAdd(&agg[(size_t)d * GCN + c], x[(size_t)s * GCN + c]);
}

__global__ void deg_count(const int* __restrict__ dst, float* __restrict__ deg) {
  int e = blockIdx.x * blockDim.x + threadIdx.x;
  if (e < NEDGE) atomicAdd(&deg[dst[e]], 1.0f);
}

__global__ void deg_finalize(float* __restrict__ deg) {
  int n = blockIdx.x * blockDim.x + threadIdx.x;
  if (n < NNODE) deg[n] = 1.0f / fmaxf(deg[n], 1.0f);
}

// bank = concat(node_feat, h1, h2)  [NNODE, 1536] fp32
__global__ void build_bank(const float* __restrict__ nf, const float* __restrict__ h1,
                           const float* __restrict__ h2, float* __restrict__ bank)
{
  int i = blockIdx.x * blockDim.x + threadIdx.x;
  if (i >= NNODE * BANKD) return;
  int n = i / BANKD, c = i % BANKD;
  float v;
  if (c < GCN)          v = nf[(size_t)n * GCN + c];
  else if (c < 2 * GCN) v = h1[(size_t)n * GCN + (c - GCN)];
  else                  v = h2[(size_t)n * GCN + (c - 2 * GCN)];
  bank[i] = v;
}

// head/tail indices with the +(==0)-1 fixup, as global node ids
__global__ void pair_idx(const int* __restrict__ pairs, int* __restrict__ HI,
                         int* __restrict__ TI)
{
  int r = blockIdx.x * blockDim.x + threadIdx.x;
  if (r >= NPAIRTOT) return;
  int b = r >> 7;
  int h = pairs[r * 2 + 0];
  int t = pairs[r * 2 + 1];
  h = h + (h == 0 ? 1 : 0) - 1;
  t = t + (t == 0 ? 1 : 0) - 1;
  HI[r] = b * 128 + h;
  TI[r] = b * 128 + t;
}

// out[r, ooff:ooff+cols] = bf16(tbl[idx[r], 0:cols])
__global__ void gather_rows_bf(const float* __restrict__ tbl, const int* __restrict__ idx,
                               __bf16* __restrict__ out, int rows, int cols,
                               int ostride, int ooff)
{
  int i = blockIdx.x * blockDim.x + threadIdx.x;
  if (i >= rows * cols) return;
  int r = i / cols, c = i % cols;
  out[(size_t)r * ostride + ooff + c] = f2bf(tbl[(size_t)idx[r] * cols + c]);
}

// XM[row, :] = bf16(h2[b*128 + path_mid[row], :])
__global__ void gather_mid_bf(const float* __restrict__ h2, const int* __restrict__ pmid,
                              __bf16* __restrict__ XM)
{
  int i = blockIdx.x * blockDim.x + threadIdx.x;
  if (i >= PATHROWS * GCN) return;
  int row = i >> 9, c = i & 511;
  int b = row >> 10;
  int m = pmid[row];
  XM[i] = f2bf(h2[(size_t)(b * 128 + m) * GCN + c]);
}

// path_cat[row, 4*GCN] (bf16) = concat of 4 tanh edge features
__global__ void combine_edge(const float* __restrict__ XHa, const float* __restrict__ XHb,
                             const float* __restrict__ XTa, const float* __restrict__ XTb,
                             const float* __restrict__ XMa, const float* __restrict__ XMb,
                             const float* __restrict__ be, __bf16* __restrict__ pc)
{
  int i = blockIdx.x * blockDim.x + threadIdx.x;
  if (i >= PATHROWS * GCN) return;
  int row = i >> 9, c = i & 511;
  int bp = row >> 3;
  float b  = be[c];
  float ha = XHa[(size_t)bp * GCN + c], hb = XHb[(size_t)bp * GCN + c];
  float ta = XTa[(size_t)bp * GCN + c], tb = XTb[(size_t)bp * GCN + c];
  float ma = XMa[(size_t)row * GCN + c], mb = XMb[(size_t)row * GCN + c];
  size_t base = (size_t)row * FOURGCN;
  pc[base + c]           = f2bf(tanhf(ha + mb + b));  // edge_feat(xh, xm)
  pc[base + GCN + c]     = f2bf(tanhf(ma + tb + b));  // edge_feat(xm, xt)
  pc[base + 2 * GCN + c] = f2bf(tanhf(ta + mb + b));  // edge_feat(xt, xm)
  pc[base + 3 * GCN + c] = f2bf(tanhf(ma + hb + b));  // edge_feat(xm, xh)
}

// ---------------------------------------------------------------------------
// Bilinear attention over K=8 paths: score -> softmax -> weighted sum.
// ---------------------------------------------------------------------------
__global__ __launch_bounds__(256) void attn_kernel(
    const float* __restrict__ qW, const float* __restrict__ path,
    const float* __restrict__ b_bil, float* __restrict__ pinfo)
{
  __shared__ float red[256];
  __shared__ float sw[KPATH];
  int bp = blockIdx.x;
  int tid = threadIdx.x;
  int k = tid >> 5, lane = tid & 31;

  const float* pr = path + (size_t)(bp * KPATH + k) * FOURGCN;
  const float* qr = qW + (size_t)bp * FOURGCN;
  float s = 0.0f;
  for (int c = lane; c < FOURGCN; c += 32) s += qr[c] * pr[c];
  red[tid] = s;
  __syncthreads();
  for (int off = 16; off > 0; off >>= 1) {
    if (lane < off) red[tid] += red[tid + off];
    __syncthreads();
  }
  if (lane == 0) sw[k] = red[tid] + b_bil[0];
  __syncthreads();

  float mx = sw[0];
#pragma unroll
  for (int i = 1; i < KPATH; ++i) mx = fmaxf(mx, sw[i]);
  float w[KPATH], sum = 0.0f;
#pragma unroll
  for (int i = 0; i < KPATH; ++i) { w[i] = __expf(sw[i] - mx); sum += w[i]; }
  float inv = 1.0f / sum;

  for (int c = tid; c < FOURGCN; c += 256) {
    float acc = 0.0f;
#pragma unroll
    for (int i = 0; i < KPATH; ++i)
      acc += w[i] * path[(size_t)(bp * KPATH + i) * FOURGCN + c];
    pinfo[(size_t)bp * FOURGCN + c] = acc * inv;
  }
}

// feat (bf16) = concat(h_ent, t_ent, |h-t|, h*t, global_info, path_info)
__global__ void build_feat(const float* __restrict__ bank, const int* __restrict__ HI,
                           const int* __restrict__ TI, const float* __restrict__ gi,
                           const float* __restrict__ pinfo, __bf16* __restrict__ feat)
{
  int i = blockIdx.x * blockDim.x + threadIdx.x;
  if (i >= NPAIRTOT * FEATDIM) return;
  int r = i / FEATDIM, c = i % FEATDIM;
  int b = r >> 7;
  float v;
  if (c < BANKD)          v = bank[(size_t)HI[r] * BANKD + c];
  else if (c < 2 * BANKD) v = bank[(size_t)TI[r] * BANKD + (c - BANKD)];
  else if (c < 3 * BANKD) {
    int cc = c - 2 * BANKD;
    v = fabsf(bank[(size_t)HI[r] * BANKD + cc] - bank[(size_t)TI[r] * BANKD + cc]);
  } else if (c < 4 * BANKD) {
    int cc = c - 3 * BANKD;
    v = bank[(size_t)HI[r] * BANKD + cc] * bank[(size_t)TI[r] * BANKD + cc];
  } else if (c < 4 * BANKD + GCN) v = gi[(size_t)b * GCN + (c - 4 * BANKD)];
  else                            v = pinfo[(size_t)r * FOURGCN + (c - 4 * BANKD - GCN)];
  feat[i] = f2bf(v);
}

// ---------------------------------------------------------------------------
// Host-side launch
// ---------------------------------------------------------------------------
extern "C" void kernel_launch(void* const* d_in, const int* in_sizes, int n_in,
                              void* d_out, int out_size, void* d_ws, size_t ws_size,
                              hipStream_t stream)
{
  const float* node_feat   = (const float*)d_in[0];
  const float* global_info = (const float*)d_in[1];
  const float* W_gcn1      = (const float*)d_in[2];
  const float* b_gcn1      = (const float*)d_in[3];
  const float* W_gcn2      = (const float*)d_in[4];
  const float* b_gcn2      = (const float*)d_in[5];
  const float* W_edge      = (const float*)d_in[6];
  const float* b_edge      = (const float*)d_in[7];
  const float* W_pm        = (const float*)d_in[8];
  const float* b_pm        = (const float*)d_in[9];
  const float* W_bil       = (const float*)d_in[10];
  const float* b_bil       = (const float*)d_in[11];
  const float* W_p1        = (const float*)d_in[12];
  const float* b_p1        = (const float*)d_in[13];
  const float* W_p2        = (const float*)d_in[14];
  const float* b_p2        = (const float*)d_in[15];
  const int*   edge_src    = (const int*)d_in[16];
  const int*   edge_dst    = (const int*)d_in[17];
  const int*   h_t_pairs   = (const int*)d_in[18];
  const int*   path_mid    = (const int*)d_in[19];
  float*       out         = (float*)d_out;

  char* ws = (char*)d_ws;
  size_t off = 0;
  auto alloc = [&](size_t bytes) -> void* {
    void* p = ws + off;
    off += (bytes + 255) & ~(size_t)255;
    return p;
  };
  // fp32 intermediates
  float* deg    = (float*)alloc((size_t)NNODE * 4);
  float* agg    = (float*)alloc((size_t)NNODE * GCN * 4);
  float* h1     = (float*)alloc((size_t)NNODE * GCN * 4);
  float* h2     = (float*)alloc((size_t)NNODE * GCN * 4);
  float* bank   = (float*)alloc((size_t)NNODE * BANKD * 4);
  int*   HI     = (int*)alloc((size_t)NPAIRTOT * 4);
  int*   TI     = (int*)alloc((size_t)NPAIRTOT * 4);
  float* XHa    = (float*)alloc((size_t)NPAIRTOT * GCN * 4);
  float* XHb    = (float*)alloc((size_t)NPAIRTOT * GCN * 4);
  float* XTa    = (float*)alloc((size_t)NPAIRTOT * GCN * 4);
  float* XTb    = (float*)alloc((size_t)NPAIRTOT * GCN * 4);
  float* XMa    = (float*)alloc((size_t)PATHROWS * GCN * 4);
  float* XMb    = (float*)alloc((size_t)PATHROWS * GCN * 4);
  float* path   = (float*)alloc((size_t)PATHROWS * FOURGCN * 4);
  float* qW     = (float*)alloc((size_t)NPAIRTOT * FOURGCN * 4);
  float* pinfo  = (float*)alloc((size_t)NPAIRTOT * FOURGCN * 4);
  // bf16 GEMM operands
  __bf16* aggbf = (__bf16*)alloc((size_t)NNODE * GCN * 2);
  __bf16* XHbf  = (__bf16*)alloc((size_t)NPAIRTOT * GCN * 2);
  __bf16* XTbf  = (__bf16*)alloc((size_t)NPAIRTOT * GCN * 2);
  __bf16* XMbf  = (__bf16*)alloc((size_t)PATHROWS * GCN * 2);
  __bf16* pcbf  = (__bf16*)alloc((size_t)PATHROWS * FOURGCN * 2);
  __bf16* Qbf   = (__bf16*)alloc((size_t)NPAIRTOT * QDIM * 2);
  __bf16* featb = (__bf16*)alloc((size_t)NPAIRTOT * FEATDIM * 2);
  __bf16* hidbf = (__bf16*)alloc((size_t)NPAIRTOT * HIDDIM * 2);
  // bf16 transposed weights [N,K]
  __bf16* WTg1  = (__bf16*)alloc((size_t)GCN * GCN * 2);
  __bf16* WTg2  = (__bf16*)alloc((size_t)GCN * GCN * 2);
  __bf16* WTea  = (__bf16*)alloc((size_t)GCN * GCN * 2);
  __bf16* WTeb  = (__bf16*)alloc((size_t)GCN * GCN * 2);
  __bf16* WTpm  = (__bf16*)alloc((size_t)FOURGCN * FOURGCN * 2);
  __bf16* WTbil = (__bf16*)alloc((size_t)QDIM * FOURGCN * 2);
  __bf16* WTp1  = (__bf16*)alloc((size_t)FEATDIM * HIDDIM * 2);
  __bf16* WTp2  = (__bf16*)alloc((size_t)HIDDIM * OUTDIM * 2);

  auto gemm = [&](const __bf16* A, const __bf16* Bt, const float* bias,
                  const float* rowscale, void* C, int M, int N, int K,
                  int ftanh, int obf) {
    dim3 g((N + BN - 1) / BN, (M + BM - 1) / BM);
    if (N % BN == 0)
      gemm_wmma<false><<<g, dim3(256), 0, stream>>>(A, Bt, bias, rowscale, C,
                                                    M, N, K, ftanh, obf);
    else
      gemm_wmma<true><<<g, dim3(256), 0, stream>>>(A, Bt, bias, rowscale, C,
                                                   M, N, K, ftanh, obf);
  };
  auto cvt_t = [&](const float* W, __bf16* Wt, int K, int N) {
    convert_transpose<<<((size_t)K * N + 255) / 256, 256, 0, stream>>>(W, Wt, K, N);
  };

  // ---- one-time (per-call) weight conversion + transpose ----
  cvt_t(W_gcn1, WTg1, GCN, GCN);
  cvt_t(W_gcn2, WTg2, GCN, GCN);
  cvt_t(W_edge, WTea, GCN, GCN);                       // W_edge[:GCN]
  cvt_t(W_edge + (size_t)GCN * GCN, WTeb, GCN, GCN);   // W_edge[GCN:]
  cvt_t(W_pm, WTpm, FOURGCN, FOURGCN);
  cvt_t(W_bil, WTbil, QDIM, FOURGCN);
  cvt_t(W_p1, WTp1, FEATDIM, HIDDIM);
  cvt_t(W_p2, WTp2, HIDDIM, OUTDIM);

  // ---- degree (shared by both GCN layers) ----
  hipMemsetAsync(deg, 0, (size_t)NNODE * 4, stream);
  deg_count<<<(NEDGE + 255) / 256, 256, 0, stream>>>(edge_dst, deg);
  deg_finalize<<<(NNODE + 255) / 256, 256, 0, stream>>>(deg);

  // ---- GCN layer 1 ----
  hipMemsetAsync(agg, 0, (size_t)NNODE * GCN * 4, stream);
  spmm_atomic<<<NEDGE * 2, 256, 0, stream>>>(node_feat, edge_src, edge_dst, agg);
  convert_bf<<<(NNODE * GCN + 255) / 256, 256, 0, stream>>>(agg, aggbf, NNODE * GCN);
  gemm(aggbf, WTg1, b_gcn1, deg, h1, NNODE, GCN, GCN, 1, 0);

  // ---- GCN layer 2 ----
  hipMemsetAsync(agg, 0, (size_t)NNODE * GCN * 4, stream);
  spmm_atomic<<<NEDGE * 2, 256, 0, stream>>>(h1, edge_src, edge_dst, agg);
  convert_bf<<<(NNODE * GCN + 255) / 256, 256, 0, stream>>>(agg, aggbf, NNODE * GCN);
  gemm(aggbf, WTg2, b_gcn2, deg, h2, NNODE, GCN, GCN, 1, 0);

  // ---- bank + pair indices ----
  build_bank<<<(NNODE * BANKD + 255) / 256, 256, 0, stream>>>(node_feat, h1, h2, bank);
  pair_idx<<<(NPAIRTOT + 255) / 256, 256, 0, stream>>>(h_t_pairs, HI, TI);

  // ---- gather xh / xt / xm (bf16) from x_fin = h2 ----
  gather_rows_bf<<<(NPAIRTOT * GCN + 255) / 256, 256, 0, stream>>>(h2, HI, XHbf, NPAIRTOT, GCN, GCN, 0);
  gather_rows_bf<<<(NPAIRTOT * GCN + 255) / 256, 256, 0, stream>>>(h2, TI, XTbf, NPAIRTOT, GCN, GCN, 0);
  gather_mid_bf<<<(PATHROWS * GCN + 255) / 256, 256, 0, stream>>>(h2, path_mid, XMbf);

  // ---- edge-feature projections (6 GEMMs) ----
  gemm(XHbf, WTea, nullptr, nullptr, XHa, NPAIRTOT, GCN, GCN, 0, 0);
  gemm(XHbf, WTeb, nullptr, nullptr, XHb, NPAIRTOT, GCN, GCN, 0, 0);
  gemm(XTbf, WTea, nullptr, nullptr, XTa, NPAIRTOT, GCN, GCN, 0, 0);
  gemm(XTbf, WTeb, nullptr, nullptr, XTb, NPAIRTOT, GCN, GCN, 0, 0);
  gemm(XMbf, WTea, nullptr, nullptr, XMa, PATHROWS, GCN, GCN, 0, 0);
  gemm(XMbf, WTeb, nullptr, nullptr, XMb, PATHROWS, GCN, GCN, 0, 0);

  // ---- combine + path mapping ----
  combine_edge<<<(PATHROWS * GCN + 255) / 256, 256, 0, stream>>>(
      XHa, XHb, XTa, XTb, XMa, XMb, b_edge, pcbf);
  gemm(pcbf, WTpm, b_pm, nullptr, path, PATHROWS, FOURGCN, FOURGCN, 0, 0);

  // ---- bilinear attention ----
  gather_rows_bf<<<(NPAIRTOT * BANKD + 255) / 256, 256, 0, stream>>>(bank, HI, Qbf, NPAIRTOT, BANKD, QDIM, 0);
  gather_rows_bf<<<(NPAIRTOT * BANKD + 255) / 256, 256, 0, stream>>>(bank, TI, Qbf, NPAIRTOT, BANKD, QDIM, BANKD);
  gemm(Qbf, WTbil, nullptr, nullptr, qW, NPAIRTOT, FOURGCN, QDIM, 0, 0);
  attn_kernel<<<NPAIRTOT, 256, 0, stream>>>(qW, path, b_bil, pinfo);

  // ---- classifier ----
  build_feat<<<(NPAIRTOT * FEATDIM + 255) / 256, 256, 0, stream>>>(
      bank, HI, TI, global_info, pinfo, featb);
  gemm(featb, WTp1, b_p1, nullptr, hidbf, NPAIRTOT, HIDDIM, FEATDIM, 1, 1);
  gemm(hidbf, WTp2, b_p2, nullptr, out, NPAIRTOT, OUTDIM, HIDDIM, 0, 0);

  (void)in_sizes; (void)n_in; (void)out_size; (void)ws_size;
}
